// CatalanPyramid_69372311765516
// MI455X (gfx1250) — matrix operand order repaired and checked by reading, not compiled, not run on hardware
//
#include <hip/hip_runtime.h>

// ---------------------------------------------------------------------------
// Gumbel Tree-LSTM (Catalan pyramid), fully fused: one workgroup per batch
// element. All state (h,c), staged f16 weights, and per-iteration GEMM output
// live in LDS. Both GEMMs use v_wmma_f32_16x16x32_f16 (fp32 accumulate).
// All WMMA operand fetches AND stores are unconditional, in-bounds LDS/global
// accesses (zero-padded buffers) to avoid per-element exec-mask toggling.
// ---------------------------------------------------------------------------

typedef __attribute__((ext_vector_type(16))) _Float16 v16h;
typedef __attribute__((ext_vector_type(8)))  float    v8f;

#define B_    1024
#define L_    64
#define HID_  512
#define LOW_  20
#define NC_   100   // 5*LOW
#define NCP_  112   // N padded to 7*16
#define KC_   40    // 2*LOW
#define HPAD  44    // hbuf column padding: covers k-20 up to 43 (k<=63), zeros
#define EPS_  1e-20f
#define NTHREADS 128
#define MAXPT 10    // ceil(63*20 / 128)

__device__ __forceinline__ float sigf(float x) { return 1.0f / (1.0f + __expf(-x)); }

__global__ __launch_bounds__(NTHREADS)
void catalan_pyramid_kernel(const float* __restrict__ input_h,   // [B][64][512]
                            const float* __restrict__ input_c,   // [B][64][512]
                            const float* __restrict__ W_reduce,  // [512][20]
                            const float* __restrict__ b_reduce,  // [20]
                            const float* __restrict__ W_comp,    // [40][100]
                            const float* __restrict__ b_comp,    // [100]
                            const float* __restrict__ query,     // [20]
                            const float* __restrict__ u_noise,   // [63][B][63]
                            const int*   __restrict__ length,    // [B]
                            float* __restrict__ out)             // [B][20]
{
    // Wr (phase-1 B staging, zero-padded to 32 cols) and V (tree-phase GEMM
    // out, padded to 64 rows so C stores are unconditional; also holds
    // nh (cols 0..19) / nc (cols 20..39) after gates) never coexist.
    __shared__ union {
        _Float16 Wr[32][HID_];     // [n][k], zero for n >= 20        (32 KB)
        float    V[L_][NCP_];      // [m][n], rows >= rows are junk   (28.7 KB)
    } u;
    __shared__ _Float16 Wc[NCP_][64];    // [n][k], zero-padded       (14 KB)
    __shared__ float hbuf[L_ + 1][HPAD]; // cols 20..43 stay zero     (11.4 KB)
    __shared__ float cbuf[L_][LOW_];
    __shared__ float zbuf[L_];
    __shared__ float bc[NCP_];
    __shared__ float br[32];             // b_reduce zero-padded
    __shared__ float qv[LOW_];
    __shared__ int   s_sel;
    __shared__ int   s_done;

    const int b    = blockIdx.x;
    const int tid  = threadIdx.x;
    const int lane = tid & 31;
    const int wave = tid >> 5;
    const int len  = length[b];

    const int ml   = lane & 15;        // M (A/C) or N (B/C) sub-index
    const int hi   = lane >> 4;        // half-wave select
    const int kb_a = hi ? 8 : 0;       // unused; A and B fetches both use kb_b
    const int kb_b = hi ? 16 : 0;      // K base for A and B (contiguous k = 16*hi + s)

    // ---------------- stage weights into LDS (f16), zero hbuf ----------------
    for (int idx = tid; idx < 32 * HID_; idx += NTHREADS) {
        int n = idx / HID_, k = idx % HID_;
        u.Wr[n][k] = (n < LOW_) ? (_Float16)W_reduce[k * LOW_ + n] : (_Float16)0.0f;
    }
    for (int idx = tid; idx < NCP_ * 64; idx += NTHREADS) {
        int n = idx / 64, k = idx % 64;
        float v = (n < NC_ && k < KC_) ? W_comp[k * NC_ + n] : 0.0f;
        Wc[n][k] = (_Float16)v;
    }
    for (int idx = tid; idx < (L_ + 1) * HPAD; idx += NTHREADS)
        (&hbuf[0][0])[idx] = 0.0f;
    for (int idx = tid; idx < NCP_; idx += NTHREADS) bc[idx] = (idx < NC_) ? b_comp[idx] : 0.0f;
    if (tid < 32)   br[tid] = (tid < LOW_) ? b_reduce[tid] : 0.0f;
    if (tid < LOW_) qv[tid] = query[tid];
    __syncthreads();

    // ---------------- phase 1: h = X@W_reduce + b (and c), WMMA ----------------
    // 8 output tiles: {h,c} x 4 row-tiles of 16; each of 4 waves does 2 tiles.
    for (int tt = wave; tt < 8; tt += 4) {
        const int mat   = tt >> 2;     // 0 = h, 1 = c
        const int mtile = tt & 3;
        const float* src =
            (mat ? input_c : input_h) + ((size_t)b * L_ + mtile * 16 + ml) * HID_;
        v8f acc0 = {}; v8f acc1 = {};
        for (int s = 0; s < HID_ / 32; ++s) {
            v16h a;
            #pragma unroll
            for (int v = 0; v < 8; ++v) {
                int koff = kb_b + 2 * v;
                float2 p = *(const float2*)(src + s * 32 + koff);
                a[2 * v]     = (_Float16)p.x;
                a[2 * v + 1] = (_Float16)p.y;
            }
            v16h b0, b1;
            #pragma unroll
            for (int v = 0; v < 8; ++v) {
                int k = s * 32 + kb_b + 2 * v;
                b0[2 * v]     = u.Wr[ml][k];
                b0[2 * v + 1] = u.Wr[ml][k + 1];
                b1[2 * v]     = u.Wr[16 + ml][k];       // zero rows for n>=20
                b1[2 * v + 1] = u.Wr[16 + ml][k + 1];
            }
            acc0 = __builtin_amdgcn_wmma_f32_16x16x32_f16(false, a, false, b0,
                                                          (short)0, acc0, false, false);
            acc1 = __builtin_amdgcn_wmma_f32_16x16x32_f16(false, a, false, b1,
                                                          (short)0, acc1, false, false);
        }
        // Store epilogue: branch-free base/stride select, bias from LDS,
        // single exec toggle for the n>=16 columns (ml < 4 lanes only).
        float* dst       = mat ? &cbuf[0][0] : &hbuf[0][0];
        const int stride = mat ? LOW_ : HPAD;
        const int mbase  = mtile * 16 + (hi ? 8 : 0);
        const float bias0 = br[ml];
        #pragma unroll
        for (int r = 0; r < 8; ++r)
            dst[(mbase + r) * stride + ml] = acc0[r] + bias0;
        if (ml < 4) {
            const int   n1    = 16 + ml;
            const float bias1 = br[n1];
            #pragma unroll
            for (int r = 0; r < 8; ++r)
                dst[(mbase + r) * stride + n1] = acc1[r] + bias1;
        }
    }
    __syncthreads();

    // ---------------- sequential pyramid ----------------
    for (int i = 0; i < L_ - 1; ++i) {
        const int rows = L_ - 1 - i;

        // --- V = [h_l | h_r] @ W_comp + b_comp via WMMA; wave w = M-tile w ---
        const int mtile = wave;
        if (mtile * 16 < rows) {
            const float* base = &hbuf[mtile * 16 + ml][0];
            v16h a0, a1;   // K-steps 0 (K=0..31) and 1 (K=32..63; zeros pad)
            #pragma unroll
            for (int v = 0; v < 8; ++v) {
                int koff = kb_b + 2 * v;
                // X[m][k] = k<20 ? h[m][k] : h[m+1][k-20]; h[m+1][j]=base[44+j]
                int o0 = (koff < LOW_) ? koff : (koff + HPAD - LOW_);
                float2 p0 = *(const float2*)(base + o0);
                float2 p1 = *(const float2*)(base + koff + 32 + HPAD - LOW_);
                a0[2 * v]     = (_Float16)p0.x;
                a0[2 * v + 1] = (_Float16)p0.y;
                a1[2 * v]     = (_Float16)p1.x;
                a1[2 * v + 1] = (_Float16)p1.y;
            }
            const int mm = mtile * 16 + (hi ? 8 : 0);
            for (int nt = 0; nt < 7; ++nt) {
                const int n = nt * 16 + ml;
                v16h b0, b1;
                #pragma unroll
                for (int v = 0; v < 8; ++v) {
                    int k = kb_b + 2 * v;
                    b0[2 * v]     = Wc[n][k];
                    b0[2 * v + 1] = Wc[n][k + 1];
                    b1[2 * v]     = Wc[n][32 + k];
                    b1[2 * v + 1] = Wc[n][32 + k + 1];
                }
                v8f acc = {};
                acc = __builtin_amdgcn_wmma_f32_16x16x32_f16(false, a0, false, b0,
                                                             (short)0, acc, false, false);
                acc = __builtin_amdgcn_wmma_f32_16x16x32_f16(false, a1, false, b1,
                                                             (short)0, acc, false, false);
                const float bn = bc[n];
                #pragma unroll
                for (int r = 0; r < 8; ++r)           // V padded to 64 rows:
                    u.V[mm + r][n] = acc[r] + bn;     // unconditional stores
            }
        }
        __syncthreads();

        // --- TreeLSTM gates; write nh -> V[:,0..19], nc -> V[:,20..39] ---
        {
            float gnh[MAXPT], gnc[MAXPT];
            int cnt = 0;
            for (int idx = tid; idx < rows * LOW_; idx += NTHREADS) {
                int j = idx / LOW_, d = idx % LOW_;
                float i_ = u.V[j][d];
                float fl = u.V[j][d + 20];
                float fr = u.V[j][d + 40];
                float uu = u.V[j][d + 60];
                float o_ = u.V[j][d + 80];
                float cl = cbuf[j][d], cr = cbuf[j + 1][d];
                float ncv = cl * sigf(fl + 1.0f) + cr * sigf(fr + 1.0f) + tanhf(uu) * sigf(i_);
                gnh[cnt] = sigf(o_) * tanhf(ncv);
                gnc[cnt] = ncv;
                ++cnt;
            }
            __syncthreads();
            cnt = 0;
            for (int idx = tid; idx < rows * LOW_; idx += NTHREADS) {
                int j = idx / LOW_, d = idx % LOW_;
                u.V[j][d]      = gnh[cnt];
                u.V[j][d + 20] = gnc[cnt];
                ++cnt;
            }
            __syncthreads();
        }

        // --- straight-through Gumbel selection (one-hot argmax) ---
        if (i < L_ - 2) {
            for (int j = tid; j < rows; j += NTHREADS) {
                float lg = 0.0f;
                #pragma unroll
                for (int d = 0; d < LOW_; ++d) lg += u.V[j][d] * qv[d];
                float uu = u_noise[((size_t)i * B_ + b) * (L_ - 1) + j];
                float g  = -__logf(-__logf(uu + EPS_) + EPS_);
                zbuf[j]  = lg + g;                       // TEMP == 1
            }
            __syncthreads();
            if (tid == 0) {
                float mz = -INFINITY;
                for (int j = 0; j < rows; ++j) mz = fmaxf(mz, zbuf[j]);
                int best = 0; float bv = -1.0f;          // first-max tie-break (jnp.argmax)
                for (int j = 0; j < rows; ++j) {
                    float sc = ((i + 1 + j) < len) ? __expf(zbuf[j] - mz) : 0.0f;
                    if (sc > bv) { bv = sc; best = j; }
                }
                s_sel  = best;
                s_done = (i + 1) < len;
            }
            __syncthreads();
        } else {
            if (tid == 0) { s_sel = 0; s_done = (i + 1) < len; }
            __syncthreads();
        }

        // --- merge at position k: h[j<k]=h[j], h[k]=nh[k], h[j>k]=h[j+1] ---
        if (s_done) {
            const int k = s_sel;
            float th[MAXPT], tc[MAXPT];
            int cnt = 0;
            for (int idx = tid; idx < rows * LOW_; idx += NTHREADS) {
                int j = idx / LOW_, d = idx % LOW_;
                // all loads unconditional + value selects (no predicated LDS)
                int jj = j + (j > k);                 // skip merged-away slot
                float hv  = hbuf[jj][d];
                float cv  = cbuf[jj][d];
                float nhv = u.V[j][d];
                float ncv = u.V[j][d + 20];
                th[cnt] = (j == k) ? nhv : hv;
                tc[cnt] = (j == k) ? ncv : cv;
                ++cnt;
            }
            __syncthreads();
            cnt = 0;
            for (int idx = tid; idx < rows * LOW_; idx += NTHREADS) {
                int j = idx / LOW_, d = idx % LOW_;
                hbuf[j][d] = th[cnt]; cbuf[j][d] = tc[cnt]; ++cnt;
            }
            __syncthreads();
        }
        // if !done: state unchanged (h = hl), effective length just shrinks.
    }

    if (tid < LOW_) out[(size_t)b * LOW_ + tid] = hbuf[0][tid];
}

extern "C" void kernel_launch(void* const* d_in, const int* in_sizes, int n_in,
                              void* d_out, int out_size, void* d_ws, size_t ws_size,
                              hipStream_t stream) {
    (void)in_sizes; (void)n_in; (void)d_ws; (void)ws_size; (void)out_size;
    const float* input_h  = (const float*)d_in[0];
    const float* input_c  = (const float*)d_in[1];
    const float* W_reduce = (const float*)d_in[2];
    const float* b_reduce = (const float*)d_in[3];
    const float* W_comp   = (const float*)d_in[4];
    const float* b_comp   = (const float*)d_in[5];
    const float* query    = (const float*)d_in[6];
    const float* u_noise  = (const float*)d_in[7];
    const int*   length   = (const int*)d_in[8];
    float* out = (float*)d_out;

    catalan_pyramid_kernel<<<B_, NTHREADS, 0, stream>>>(
        input_h, input_c, W_reduce, b_reduce, W_comp, b_comp, query, u_noise,
        length, out);
}
